// Mamba_74079595922079
// MI455X (gfx1250) — matrix operand order repaired and checked by reading, not compiled, not run on hardware
//
#include <hip/hip_runtime.h>
#include <hip/hip_bf16.h>

typedef __attribute__((ext_vector_type(16))) __bf16 v16bf;
typedef __attribute__((ext_vector_type(8)))  float  v8f;

#define DIM     192
#define DINNER  384
#define DPROJ   768
#define DSTATE  16
#define SEQ     16384
#define CHUNK   128

// ---------- helpers ----------

__device__ __forceinline__ unsigned short f2bf(float x) {
  unsigned int u = __float_as_uint(x);
  u += 0x7FFFu + ((u >> 16) & 1u);           // round-to-nearest-even
  return (unsigned short)(u >> 16);
}

// A-fragment (16x32 bf16) from LDS, row-major with stride `ld` (ushorts).
// ISA 7.12.2 16-bit A layout: lane m=lane&15; lanes>=16 shifted by 8 in K;
// VGPR pairs are contiguous K values -> two 16B LDS loads per lane.
__device__ __forceinline__ v16bf load_frag_a_lds(const unsigned short* base, int ld,
                                                 int lane, int kbase) {
  const int m  = lane & 15;
  const int h8 = ((lane >> 4) & 1) * 8;
  const unsigned short* row = base + m * ld + kbase + h8;
  v16bf f;
  ((int4*)&f)[0] = *(const int4*)(row);        // K pairs 0..3  (kbase+h8 .. +7)
  ((int4*)&f)[1] = *(const int4*)(row + 16);   // K pairs 4..7  (kbase+16+h8 ..)
  return f;
}

// B-fragment (32x16 bf16) for C = A * W^T : B[k][n] = W[n][k].
// Lane's column n = ntile*16 + (lane&15); lanes 0-15 take K=kbase..+7 and kbase+16..+23,
// lanes 16-31 take K=kbase+8..+15 and kbase+24..+31 (same order as A fragment).
__device__ __forceinline__ v16bf load_frag_b(const unsigned short* w, int ld,
                                             int lane, int ntile, int kbase) {
  const int n  = ntile * 16 + (lane & 15);
  const int kb = kbase + ((lane >> 4) & 1) * 8;
  const unsigned short* p = w + n * ld + kb;
  v16bf f;
  ((int4*)&f)[0] = *(const int4*)(p);
  ((int4*)&f)[1] = *(const int4*)(p + 16);
  return f;
}

// ---------- kernel 0: convert weights f32 -> bf16 ----------

__global__ __launch_bounds__(256) void k_cvt(const float* __restrict__ win,
                                             const float* __restrict__ wout,
                                             unsigned short* __restrict__ bin,
                                             unsigned short* __restrict__ bout) {
  int i = blockIdx.x * 256 + threadIdx.x;
  if (i < DPROJ * DIM)  bin[i]  = f2bf(win[i]);     // 147456
  if (i < DIM * DINNER) bout[i] = f2bf(wout[i]);    // 73728
}

// ---------- kernel 1: LayerNorm + in-proj GEMM (bf16 WMMA) + SiLU gate ----------
// Block = 16 rows x 768 cols, 256 threads (8 waves x 6 col-tiles), K=192 (6 steps).

__global__ __launch_bounds__(256) void k_ln_proj(
    const float* __restrict__ x, const float* __restrict__ gamma,
    const float* __restrict__ beta, const unsigned short* __restrict__ wbf,
    const float* __restrict__ b_in, float* __restrict__ u) {
  __shared__ unsigned short sA[16][200];     // 16 x 192 bf16, padded, 16B-aligned rows
  __shared__ float sP[16][DPROJ];            // projection exchange buffer
  __shared__ float sred[2][16][16];
  __shared__ float smu[16], srs[16];

  const int tid = threadIdx.x;
  const int rowbase = blockIdx.x * 16;
  const int r = tid >> 4, c = tid & 15;

  // --- LayerNorm (16 threads per row) ---
  float v[12];
  float s = 0.f, ss = 0.f;
#pragma unroll
  for (int j = 0; j < 12; j++) {
    float t = x[(rowbase + r) * DIM + c + 16 * j];
    v[j] = t; s += t; ss += t * t;
  }
  sred[0][r][c] = s; sred[1][r][c] = ss;
  __syncthreads();
  if (c == 0) {
    float S = 0.f, SS = 0.f;
#pragma unroll
    for (int j = 0; j < 16; j++) { S += sred[0][r][j]; SS += sred[1][r][j]; }
    float mu  = S * (1.f / DIM);
    float var = SS * (1.f / DIM) - mu * mu;
    smu[r] = mu; srs[r] = rsqrtf(var + 1e-5f);
  }
  __syncthreads();
  {
    float mu = smu[r], rs = srs[r];
#pragma unroll
    for (int j = 0; j < 12; j++) {
      int col = c + 16 * j;
      float xn = (v[j] - mu) * rs * gamma[col] + beta[col];
      sA[r][col] = f2bf(xn);
    }
  }
  __syncthreads();

  // --- GEMM: [16 x 192] x [192 x 768] ---
  const int wave = tid >> 5, lane = tid & 31;
  const v8f zero = {0.f, 0.f, 0.f, 0.f, 0.f, 0.f, 0.f, 0.f};
  v8f acc[6];
#pragma unroll
  for (int i = 0; i < 6; i++) acc[i] = zero;

#pragma unroll
  for (int ks = 0; ks < 6; ks++) {
    v16bf af = load_frag_a_lds(&sA[0][0], 200, lane, ks * 32);
#pragma unroll
    for (int i = 0; i < 6; i++) {
      v16bf bf = load_frag_b(wbf, DIM, lane, wave * 6 + i, ks * 32);
      acc[i] = __builtin_amdgcn_wmma_f32_16x16x32_bf16(
          false, af, false, bf, (short)0, acc[i], false, false);
    }
  }

  // C layout: VGPR rr -> M = rr + (lane>=16 ? 8 : 0), N = lane&15
#pragma unroll
  for (int i = 0; i < 6; i++) {
    int n0 = (wave * 6 + i) * 16 + (lane & 15);
    float bias = b_in[n0];
#pragma unroll
    for (int rr = 0; rr < 8; rr++) {
      int mrow = rr + ((lane >> 4) << 3);
      sP[mrow][n0] = acc[i][rr] + bias;
    }
  }
  __syncthreads();

  // --- SiLU gate: u = p[:, :384] * sigmoid(p[:, 384:]) ---
#pragma unroll
  for (int j = 0; j < 24; j++) {
    int idx = tid + 256 * j;                  // 0..6143
    int rr = idx / DINNER;
    int cc = idx - rr * DINNER;
    float a = sP[rr][cc];
    float g = sP[rr][cc + DINNER];
    u[(rowbase + rr) * DINNER + cc] = a * (1.f / (1.f + __expf(-g)));
  }
}

// ---------- kernel 2: causal depthwise conv (4 taps) + chunked SSM scan ----------
// One thread per (batch, chunk, channel); chunks reset state (h0=0) per reference.

__global__ __launch_bounds__(256) void k_conv_scan(
    const float* __restrict__ u, const float* __restrict__ conv_w,
    const float* __restrict__ A, const float* __restrict__ Bp,
    const float* __restrict__ Cp, unsigned short* __restrict__ ybf) {
  int gid = blockIdx.x * 256 + threadIdx.x;
  int ch = gid % DINNER;
  int cg = gid / DINNER;            // 0..1023
  int b = cg >> 7;
  int chunk = cg & 127;
  int seqbase = chunk * CHUNK;
  int rowbase = b * SEQ + seqbase;

  float w0 = conv_w[ch * 4 + 0], w1 = conv_w[ch * 4 + 1];
  float w2 = conv_w[ch * 4 + 2], w3 = conv_w[ch * 4 + 3];

  float av[DSTATE], bv[DSTATE], cv[DSTATE], h[DSTATE];
#pragma unroll
  for (int s = 0; s < DSTATE; s++) {          // uniform -> SGPRs
    av[s] = __expf(-__expf(A[s]));
    bv[s] = Bp[s]; cv[s] = Cp[s]; h[s] = 0.f;
  }

  // conv history crosses chunk boundary (conv is over full sequence)
  float um3 = (seqbase >= 3) ? u[(rowbase - 3) * DINNER + ch] : 0.f;
  float um2 = (seqbase >= 2) ? u[(rowbase - 2) * DINNER + ch] : 0.f;
  float um1 = (seqbase >= 1) ? u[(rowbase - 1) * DINNER + ch] : 0.f;

  for (int t = 0; t < CHUNK; t++) {
    float ut = u[(rowbase + t) * DINNER + ch];
    float xc = fmaf(w0, um3, fmaf(w1, um2, fmaf(w2, um1, w3 * ut)));
    um3 = um2; um2 = um1; um1 = ut;
    float yt = 0.f;
#pragma unroll
    for (int s = 0; s < DSTATE; s++) {
      h[s] = fmaf(av[s], h[s], xc * bv[s]);
      yt = fmaf(h[s], cv[s], yt);
    }
    ybf[(rowbase + t) * DINNER + ch] = f2bf(yt);
  }
}

// ---------- kernel 3: out-proj GEMM (bf16 WMMA) + bias + D*residual ----------
// Block = 16 rows x 192 cols, 128 threads (4 waves x 3 col-tiles), K=384 (12 steps).

__global__ __launch_bounds__(128) void k_out_proj(
    const unsigned short* __restrict__ ybf, const unsigned short* __restrict__ wbf,
    const float* __restrict__ b_out, const float* __restrict__ D,
    const float* __restrict__ xres, float* __restrict__ out) {
  __shared__ unsigned short sY[16][392];      // 16 x 384 bf16, padded
  const int tid = threadIdx.x;
  const int rowbase = blockIdx.x * 16;

  // cooperative stage of y tile (already bf16 in workspace), 8B per transfer
#pragma unroll
  for (int j = 0; j < 12; j++) {
    int e = (tid + 128 * j) * 4;              // element index, 384%4==0 -> no row cross
    int rr = e / DINNER;
    int cc = e - rr * DINNER;
    *(uint2*)(&sY[rr][cc]) = *(const uint2*)(ybf + (rowbase + rr) * DINNER + cc);
  }
  __syncthreads();

  const int wave = tid >> 5, lane = tid & 31;
  const v8f zero = {0.f, 0.f, 0.f, 0.f, 0.f, 0.f, 0.f, 0.f};
  v8f acc[3];
#pragma unroll
  for (int i = 0; i < 3; i++) acc[i] = zero;

#pragma unroll
  for (int ks = 0; ks < 12; ks++) {
    v16bf af = load_frag_a_lds(&sY[0][0], 392, lane, ks * 32);
#pragma unroll
    for (int i = 0; i < 3; i++) {
      v16bf bf = load_frag_b(wbf, DINNER, lane, wave * 3 + i, ks * 32);
      acc[i] = __builtin_amdgcn_wmma_f32_16x16x32_bf16(
          false, af, false, bf, (short)0, acc[i], false, false);
    }
  }

#pragma unroll
  for (int i = 0; i < 3; i++) {
    int n0 = (wave * 3 + i) * 16 + (lane & 15);
    float bb = b_out[n0], dd = D[n0];
#pragma unroll
    for (int rr = 0; rr < 8; rr++) {
      int mrow = rr + ((lane >> 4) << 3);
      int row = rowbase + mrow;
      out[row * DIM + n0] = acc[i][rr] + bb + dd * xres[row * DIM + n0];
    }
  }
}

// ---------- launch ----------

extern "C" void kernel_launch(void* const* d_in, const int* in_sizes, int n_in,
                              void* d_out, int out_size, void* d_ws, size_t ws_size,
                              hipStream_t stream) {
  const float* x      = (const float*)d_in[0];
  const float* gamma  = (const float*)d_in[1];
  const float* beta   = (const float*)d_in[2];
  const float* w_in   = (const float*)d_in[3];
  const float* b_in   = (const float*)d_in[4];
  const float* conv_w = (const float*)d_in[5];
  const float* A      = (const float*)d_in[6];
  const float* Bp     = (const float*)d_in[7];
  const float* Cp     = (const float*)d_in[8];
  const float* D      = (const float*)d_in[9];
  const float* w_out  = (const float*)d_in[10];
  const float* b_out  = (const float*)d_in[11];
  float* out = (float*)d_out;

  // workspace layout (bytes):
  //   u (f32, 131072*384)        @ 0          : 201326592
  //   y (bf16, 131072*384)       @ 201326592  : 100663296
  //   wbf_in (bf16, 768*192)     @ 301989888  : 294912
  //   wbf_out (bf16, 192*384)    @ 302284800  : 147456
  char* ws = (char*)d_ws;
  float*          u       = (float*)(ws);
  unsigned short* ybf     = (unsigned short*)(ws + 201326592UL);
  unsigned short* wbf_in  = (unsigned short*)(ws + 301989888UL);
  unsigned short* wbf_out = (unsigned short*)(ws + 302284800UL);

  hipLaunchKernelGGL(k_cvt, dim3(576), dim3(256), 0, stream,
                     w_in, w_out, wbf_in, wbf_out);
  hipLaunchKernelGGL(k_ln_proj, dim3(8192), dim3(256), 0, stream,
                     x, gamma, beta, wbf_in, b_in, u);
  hipLaunchKernelGGL(k_conv_scan, dim3(1536), dim3(256), 0, stream,
                     u, conv_w, A, Bp, Cp, ybf);
  hipLaunchKernelGGL(k_out_proj, dim3(8192), dim3(128), 0, stream,
                     ybf, wbf_out, b_out, D, x, out);
}
